// ContextCollapseHead_47502338294478
// MI455X (gfx1250) — matrix operand order repaired and checked by reading, not compiled, not run on hardware
//
#include <hip/hip_runtime.h>

typedef __attribute__((ext_vector_type(16))) _Float16 v16h;
typedef __attribute__((ext_vector_type(2)))  _Float16 v2h;
typedef __attribute__((ext_vector_type(8)))  float    v8f;

#define WAVES 8
#define HS 260   // f32 staging row stride (floats)
#define FS 264   // f16 activation row stride (halfs)
#define XS 34
#define AS 34
// per-wave LDS floats: f32 staging + f16 acts (FS halfs = FS/2 floats) + states + attn + scalars
#define PER_WAVE_F (16*HS + 16*(FS/2) + 16*XS + 16*AS + 64)

static constexpr int Kc = 32, Dc = 128;
static constexpr int ROWS = 64 * 4096;

// packed f16 weight offsets in d_ws (units: halfs). tile = 512 halfs (32 lanes x 16)
static constexpr int OFF_ATTN_W1 = 0;        // 1 kstep x 16 ntiles
static constexpr int OFF_ATTN_W2 = 8192;     // 8 x 2
static constexpr int OFF_INTF_W1 = 16384;    // 1 x 16
static constexpr int OFF_INTF_W2 = 24576;    // 8 x 8
static constexpr int OFF_SYN_W1  = 57344;    // 3 x 16 (K=65 zero-padded to 96)
static constexpr int OFF_SYN_W2  = 81920;    // 8 x 16
static constexpr int OFF_SYN_W3  = 147456;   // 8 x 8

// ---------------- prep: pack f32 weights (Kin x Nout) into WMMA B-fragment tiles ----
__global__ void pack_b_kernel(const float* __restrict__ W, _Float16* __restrict__ dst,
                              int Kin, int Nout, int KS, int NT) {
  int idx = blockIdx.x * blockDim.x + threadIdx.x;
  int total = KS * NT * 512;
  if (idx >= total) return;
  int tile = idx >> 9;
  int e = idx & 511;
  int lane = e >> 4;          // 0..31
  int j = e & 15;             // half index within lane
  int ks = tile / NT, nt = tile % NT;
  // B (KxN) 16-bit layout: lanes 0-15 hold K=0..7,16..23 of the tile, lanes 16-31 K=8..15,24..31,
  // each lane = one column n, 16 K-values packed 2-per-VGPR.
  int k = ks * 32 + (lane >> 4) * 8 + j + (j >= 8 ? 8 : 0);
  int n = nt * 16 + (lane & 15);
  float v = (k < Kin) ? W[k * Nout + n] : 0.0f;
  dst[idx] = (_Float16)v;
}

// ---------------- wave helpers ----------------
__device__ inline float wave_sum(float v) {
#pragma unroll
  for (int m = 16; m >= 1; m >>= 1) v += __shfl_xor(v, m, 32);
  return v;
}
__device__ inline float wave_max(float v) {
#pragma unroll
  for (int m = 16; m >= 1; m >>= 1) v = fmaxf(v, __shfl_xor(v, m, 32));
  return v;
}

// Build f16 A fragments from an f32 LDS buffer (states only; needs conversion).
// ISA 16-bit A 16x32 layout: lanes 0-15 M=0..15 {v0:K0,1 .. v3:K6,7, v4:K16,17 .. v7:K22,23},
// lanes 16-31 shift K by +8.
template <int KS>
__device__ inline void build_a_f32(v16h* afr, const float* buf, int stride, int lane) {
  const int m = lane & 15, half = lane >> 4;
  const float* row = buf + m * stride + half * 8;
#pragma unroll
  for (int ks = 0; ks < KS; ++ks) {
    v16h a;
#pragma unroll
    for (int v = 0; v < 8; ++v) {
      int koff = ks * 32 + 2 * v + (v >= 4 ? 8 : 0);
      float2 p = *(const float2*)(row + koff);
      a[2 * v]     = (_Float16)p.x;
      a[2 * v + 1] = (_Float16)p.y;
    }
    afr[ks] = a;
  }
}

// Build f16 A fragments straight from packed-f16 LDS activations: pure ds_read_b32, no VALU cvt.
template <int KS>
__device__ inline void build_a_f16(v16h* afr, const _Float16* buf, int stride, int lane) {
  const int m = lane & 15, half = lane >> 4;
  const _Float16* row = buf + m * stride + half * 8;
#pragma unroll
  for (int ks = 0; ks < KS; ++ks) {
    v16h a;
#pragma unroll
    for (int v = 0; v < 8; ++v) {
      int koff = ks * 32 + 2 * v + (v >= 4 ? 8 : 0);
      v2h p = *(const v2h*)(row + koff);
      a[2 * v]     = p[0];
      a[2 * v + 1] = p[1];
    }
    afr[ks] = a;
  }
}

// GEMM over NT output tiles, KS k-steps; +bias, optional ReLU, stage f32 to LDS.
// C layout: lanes 0-15: n=lane, m=vgpr; lanes 16-31: n=lane-16, m=8+vgpr.
template <int KS>
__device__ inline void gemm_store(const v16h* afr, const _Float16* __restrict__ wp, int NT,
                                  const float* __restrict__ bias, float* obuf, int ostride,
                                  bool relu, int lane) {
  const int nlo = lane & 15;
  const int mh = (lane >> 4) * 8;
  const v16h* wv = (const v16h*)wp;  // 32-byte fragment units
  for (int nt = 0; nt < NT; ++nt) {
    v8f c = {};
#pragma unroll
    for (int ks = 0; ks < KS; ++ks) {
      v16h b = wv[(ks * NT + nt) * 32 + lane];
      c = __builtin_amdgcn_wmma_f32_16x16x32_f16(false, afr[ks], false, b,
                                                 (short)0, c, false, false);
    }
    const float bn = bias[nt * 16 + nlo];
    const int n = nt * 16 + nlo;
#pragma unroll
    for (int i = 0; i < 8; ++i) {
      float v = c[i] + bn;
      if (relu) v = fmaxf(v, 0.0f);
      obuf[(mh + i) * ostride + n] = v;
    }
  }
}

// LayerNorm over C cols of 16 rows of f32 staging; writes packed f16 activations.
// Lane pair (l, l^16) shares row l&15, each half covers C/2 cols.
__device__ inline void ln_to_f16(const float* buf, int stride, int C,
                                 _Float16* fbuf, int fstride,
                                 const float* __restrict__ gamma,
                                 const float* __restrict__ beta,
                                 bool relu, int lane) {
  const int r = lane & 15, half = lane >> 4;
  const int CN = C >> 1;
  const float* row = buf + r * stride + half * CN;
  float s = 0.0f, ss = 0.0f;
  for (int c = 0; c < CN; c += 4) {
    float4 v = *(const float4*)(row + c);
    s  += v.x + v.y + v.z + v.w;
    ss += v.x * v.x + v.y * v.y + v.z * v.z + v.w * v.w;
  }
  s  += __shfl_xor(s, 16, 32);
  ss += __shfl_xor(ss, 16, 32);
  const float mean = s / (float)C;
  const float inv = rsqrtf(ss / (float)C - mean * mean + 1e-5f);
  const int cbase = half * CN;
  _Float16* frow = fbuf + r * fstride + cbase;
  for (int c = 0; c < CN; c += 2) {
    float v0 = (row[c]     - mean) * inv * gamma[cbase + c]     + beta[cbase + c];
    float v1 = (row[c + 1] - mean) * inv * gamma[cbase + c + 1] + beta[cbase + c + 1];
    if (relu) { v0 = fmaxf(v0, 0.0f); v1 = fmaxf(v1, 0.0f); }
    v2h pv; pv[0] = (_Float16)v0; pv[1] = (_Float16)v1;
    *(v2h*)(frow + c) = pv;   // one 4-byte DS store per 2 cols
  }
}

// In-place f32 LayerNorm (final output norm, result stays f32 for the global store).
__device__ inline void ln_inplace(float* buf, int stride, int C,
                                  const float* __restrict__ gamma,
                                  const float* __restrict__ beta, int lane) {
  const int r = lane & 15, half = lane >> 4;
  const int CN = C >> 1;
  float* row = buf + r * stride + half * CN;
  float s = 0.0f, ss = 0.0f;
  for (int c = 0; c < CN; c += 4) {
    float4 v = *(const float4*)(row + c);
    s  += v.x + v.y + v.z + v.w;
    ss += v.x * v.x + v.y * v.y + v.z * v.z + v.w * v.w;
  }
  s  += __shfl_xor(s, 16, 32);
  ss += __shfl_xor(ss, 16, 32);
  const float mean = s / (float)C;
  const float inv = rsqrtf(ss / (float)C - mean * mean + 1e-5f);
  const int cbase = half * CN;
  for (int c = 0; c < CN; ++c) {
    row[c] = (row[c] - mean) * inv * gamma[cbase + c] + beta[cbase + c];
  }
}

// ---------------- main fused kernel: one wave per 16-row tile ----------------
__global__ void __launch_bounds__(WAVES * 32, 1)
cch_main(const float* __restrict__ states, const float* __restrict__ temperature,
         const _Float16* __restrict__ wpack,
         const float* __restrict__ attn_b1, const float* __restrict__ attn_g1,
         const float* __restrict__ attn_be1, const float* __restrict__ attn_b2,
         const float* __restrict__ intf_b1, const float* __restrict__ intf_g1,
         const float* __restrict__ intf_be1, const float* __restrict__ intf_b2,
         const float* __restrict__ intf_w3, const float* __restrict__ intf_b3,
         const float* __restrict__ syn_b1, const float* __restrict__ syn_g1,
         const float* __restrict__ syn_be1, const float* __restrict__ syn_b2,
         const float* __restrict__ syn_g2, const float* __restrict__ syn_be2,
         const float* __restrict__ syn_b3, const float* __restrict__ out_g,
         const float* __restrict__ out_be,
         float* __restrict__ out_p, float* __restrict__ out_c) {
  extern __shared__ float smem[];
  const int lane = threadIdx.x & 31;
  const int wave = threadIdx.x >> 5;
  float* base = smem + wave * PER_WAVE_F;
  float*     hbuf = base;                           // 16 x HS f32 (WMMA outputs / LN stats)
  _Float16*  fbuf = (_Float16*)(hbuf + 16 * HS);    // 16 x FS f16 (post-LN activations)
  float*     xbuf = (float*)(fbuf + 16 * FS);       // 16 x XS f32 (states)
  float*     abuf = xbuf + 16 * XS;                 // 16 x AS f32 (attn logits / probs)
  float*     entb = abuf + 16 * AS;                 // 16 entropy
  float*     swb  = entb + 16;                      // 16 states_weighted
  float*     cb   = swb + 16;                       // 16 contradiction

  const int tile = blockIdx.x * WAVES + wave;
  const int row0 = tile * 16;

  // ---- load states (16 x 32) ----
  for (int r = 0; r < 16; ++r)
    xbuf[r * XS + lane] = states[(row0 + r) * Kc + lane];
  __syncthreads();

  // ---- entropy of softmax(states / T), normalized by log K ----
  {
    const float invT = 1.0f / temperature[0];
    for (int r = 0; r < 16; ++r) {
      float t = xbuf[r * XS + lane] * invT;
      float mx = wave_max(t);
      float e = __expf(t - mx);
      float p = e / wave_sum(e);
      float term = wave_sum(p * __logf(p + 1e-8f));
      if (lane == 0) entb[r] = -term * 0.28853900817779268f;  // 1/ln(32)
    }
  }
  v16h afrX[1];
  build_a_f32<1>(afrX, xbuf, XS, lane);
  v16h afrH[8];
  __syncthreads();

  // ---- attention path: 32->256 (LN,ReLU) -> 32 (softmax) ----
  gemm_store<1>(afrX, wpack + OFF_ATTN_W1, 16, attn_b1, hbuf, HS, false, lane);
  __syncthreads();
  ln_to_f16(hbuf, HS, 256, fbuf, FS, attn_g1, attn_be1, true, lane);
  __syncthreads();
  build_a_f16<8>(afrH, fbuf, FS, lane);
  __syncthreads();
  gemm_store<8>(afrH, wpack + OFF_ATTN_W2, 2, attn_b2, abuf, AS, false, lane);
  __syncthreads();
  for (int r = 0; r < 16; ++r) {
    float x = abuf[r * AS + lane];
    float mx = wave_max(x);
    float e = __expf(x - mx);
    float p = e / wave_sum(e);
    abuf[r * AS + lane] = p;
    float sw = wave_sum(xbuf[r * XS + lane] * p);
    if (lane == 0) swb[r] = sw;
  }
  __syncthreads();

  // ---- interference path: 32->256 (LN,ReLU) ->128 (ReLU) ->1 (sigmoid) ----
  gemm_store<1>(afrX, wpack + OFF_INTF_W1, 16, intf_b1, hbuf, HS, false, lane);
  __syncthreads();
  ln_to_f16(hbuf, HS, 256, fbuf, FS, intf_g1, intf_be1, true, lane);
  __syncthreads();
  build_a_f16<8>(afrH, fbuf, FS, lane);
  __syncthreads();
  gemm_store<8>(afrH, wpack + OFF_INTF_W2, 8, intf_b2, hbuf, HS, true, lane);
  __syncthreads();
  {
    const float b3 = intf_b3[0];
    for (int r = 0; r < 16; ++r) {
      float acc = 0.0f;
#pragma unroll
      for (int j = 0; j < 4; ++j)
        acc += hbuf[r * HS + lane + 32 * j] * intf_w3[lane + 32 * j];
      acc = wave_sum(acc);
      if (lane == 0) {
        float sig = 1.0f / (1.0f + __expf(-(acc + b3)));
        cb[r] = 0.5f * (entb[r] + sig);
      }
    }
  }
  __syncthreads();

  // ---- synthesis input (f16): [sw x32 | attn x32 | contradiction | zero-pad to 96] ----
  for (int r = 0; r < 16; ++r) {
    _Float16* frow = fbuf + r * FS;
    frow[lane]      = (_Float16)swb[r];
    frow[32 + lane] = (_Float16)abuf[r * AS + lane];
    frow[64 + lane] = (lane == 0) ? (_Float16)cb[r] : (_Float16)0.0f;
  }
  __syncthreads();
  v16h afrS[3];
  build_a_f16<3>(afrS, fbuf, FS, lane);
  __syncthreads();
  gemm_store<3>(afrS, wpack + OFF_SYN_W1, 16, syn_b1, hbuf, HS, false, lane);
  __syncthreads();
  ln_to_f16(hbuf, HS, 256, fbuf, FS, syn_g1, syn_be1, true, lane);
  __syncthreads();
  build_a_f16<8>(afrH, fbuf, FS, lane);
  __syncthreads();
  gemm_store<8>(afrH, wpack + OFF_SYN_W2, 16, syn_b2, hbuf, HS, false, lane);
  __syncthreads();
  ln_to_f16(hbuf, HS, 256, fbuf, FS, syn_g2, syn_be2, true, lane);
  __syncthreads();
  build_a_f16<8>(afrH, fbuf, FS, lane);
  __syncthreads();
  gemm_store<8>(afrH, wpack + OFF_SYN_W3, 8, syn_b3, hbuf, HS, false, lane);
  __syncthreads();
  ln_inplace(hbuf, HS, 128, out_g, out_be, lane);
  __syncthreads();

  // ---- store p_t (16 x 128, coalesced 512B/row) and contradiction ----
  for (int r = 0; r < 16; ++r) {
    float4 v = *(const float4*)(hbuf + r * HS + lane * 4);
    *(float4*)(out_p + (size_t)(row0 + r) * Dc + lane * 4) = v;
  }
  if (lane < 16) out_c[row0 + lane] = cb[lane];
}

extern "C" void kernel_launch(void* const* d_in, const int* in_sizes, int n_in,
                              void* d_out, int out_size, void* d_ws, size_t ws_size,
                              hipStream_t stream) {
  (void)in_sizes; (void)n_in; (void)out_size; (void)ws_size;
  const float* states = (const float*)d_in[0];
  const float* temp   = (const float*)d_in[1];
  _Float16* wpack = (_Float16*)d_ws;

  auto pack = [&](int in_idx, int off, int Kin, int Nout) {
    int KS = (Kin + 31) / 32, NT = Nout / 16;
    int total = KS * NT * 512;
    pack_b_kernel<<<(total + 255) / 256, 256, 0, stream>>>(
        (const float*)d_in[in_idx], wpack + off, Kin, Nout, KS, NT);
  };
  pack(2,  OFF_ATTN_W1, 32, 256);
  pack(6,  OFF_ATTN_W2, 256, 32);
  pack(8,  OFF_INTF_W1, 32, 256);
  pack(12, OFF_INTF_W2, 256, 128);
  pack(16, OFF_SYN_W1,  65, 256);
  pack(20, OFF_SYN_W2,  256, 256);
  pack(24, OFF_SYN_W3,  256, 128);

  const int blocks = ROWS / (16 * WAVES);  // 2048
  const size_t smem = (size_t)WAVES * PER_WAVE_F * sizeof(float);  // ~232 KB
  (void)hipFuncSetAttribute((const void*)cch_main,
                            hipFuncAttributeMaxDynamicSharedMemorySize, (int)smem);

  float* out_p = (float*)d_out;
  float* out_c = out_p + (size_t)ROWS * Dc;

  cch_main<<<blocks, WAVES * 32, smem, stream>>>(
      states, temp, wpack,
      (const float*)d_in[3],  (const float*)d_in[4],  (const float*)d_in[5],
      (const float*)d_in[7],
      (const float*)d_in[9],  (const float*)d_in[10], (const float*)d_in[11],
      (const float*)d_in[13], (const float*)d_in[14], (const float*)d_in[15],
      (const float*)d_in[17], (const float*)d_in[18], (const float*)d_in[19],
      (const float*)d_in[21], (const float*)d_in[22], (const float*)d_in[23],
      (const float*)d_in[25], (const float*)d_in[26], (const float*)d_in[27],
      out_p, out_c);
}
